// DAGNet_68487548502590
// MI455X (gfx1250) — compile-verified, matrix operand order checked by reading, not run on hardware
//
#include <hip/hip_runtime.h>
#include <hip/hip_bf16.h>
#include <stdint.h>

// ---------------- sizes ----------------
#define NB      16          // batch
#define CIN     16
#define CP      32          // planes
#define HWI     128
#define HWO     64
#define NPOS    (NB*HWO*HWO)        // 65536 output positions
#define NNODES  32
#define EPSBN   1e-5f

typedef _Float16 v8h  __attribute__((ext_vector_type(8)));
typedef _Float16 v16h __attribute__((ext_vector_type(16)));
typedef float    v8f  __attribute__((ext_vector_type(8)));

struct NodeArgs { int preds[32]; int npreds; int node; };

// ---------------- prep: xr = relu(x) as f16 NHWC (16,128,128,16) ----------------
__global__ void k_prep_xr(const float* __restrict__ x, _Float16* __restrict__ xr) {
    int p = blockIdx.x * blockDim.x + threadIdx.x;      // (b,y,x) over 16*128*128
    if (p >= NB * HWI * HWI) return;
    int b = p / (HWI * HWI);
    int rem = p % (HWI * HWI);
    const float* xp = x + (size_t)b * CIN * HWI * HWI + rem;
    _Float16* o = xr + (size_t)p * CIN;
    #pragma unroll
    for (int c = 0; c < CIN; ++c) {
        float v = xp[(size_t)c * HWI * HWI];
        o[c] = (_Float16)(v > 0.f ? v : 0.f);
    }
}

// ---------------- weight packing into per-lane WMMA B-fragments ----------------
// input nodes: [ii][chunk t:0..4][ntile:0..1][lane:0..31][e:0..15]
//   chunk t covers tap pair (2t, 2t+1); element e: tap = 2t+(e>>3), ch = 8*kg+(e&7)
__global__ void k_pack_in(const float* __restrict__ w, _Float16* __restrict__ wpk, int n_in) {
    int id = blockIdx.x * blockDim.x + threadIdx.x;
    int total = n_in * 5 * 2 * 32 * 16;
    if (id >= total) return;
    int e    = id & 15;
    int lane = (id >> 4) & 31;
    int nt   = (id >> 9) & 1;
    int t    = (id >> 10) % 5;
    int ii   = id / (5 * 2 * 32 * 16);
    int kg = lane >> 4, nn = lane & 15;
    int tap = 2 * t + (e >> 3);
    int oc  = nt * 16 + nn;
    int ch  = 8 * kg + (e & 7);
    float v = 0.f;
    if (tap < 9)
        v = w[((((size_t)ii * CP + oc) * CIN + ch) * 3 + tap / 3) * 3 + tap % 3];
    wpk[id] = (_Float16)v;
}

// inner nodes: [jj][tap:0..8][ntile][lane][e];  ic = 16*(e>>3) + 8*kg + (e&7)
__global__ void k_pack_inner(const float* __restrict__ w, _Float16* __restrict__ wpk, int n_inner) {
    int id = blockIdx.x * blockDim.x + threadIdx.x;
    int total = n_inner * 9 * 2 * 32 * 16;
    if (id >= total) return;
    int e    = id & 15;
    int lane = (id >> 4) & 31;
    int nt   = (id >> 9) & 1;
    int t    = (id >> 10) % 9;
    int jj   = id / (9 * 2 * 32 * 16);
    int kg = lane >> 4, nn = lane & 15;
    int oc = nt * 16 + nn;
    int ic = 16 * (e >> 3) + 8 * kg + (e & 7);
    float v = w[((((size_t)jj * CP + oc) * CP + ic) * 3 + t / 3) * 3 + t % 3];
    wpk[id] = (_Float16)v;
}

// ---------------- gather: h = relu(sum_i sigmoid(ew[n,i]) * act[pred_i]) ----------------
__global__ void k_gather(const _Float16* __restrict__ act_base, size_t act_stride,
                         const float* __restrict__ edge_w, int max_deg,
                         NodeArgs na, _Float16* __restrict__ h) {
    int p = blockIdx.x * blockDim.x + threadIdx.x;
    if (p >= NPOS) return;
    float acc[CP];
    #pragma unroll
    for (int c = 0; c < CP; ++c) acc[c] = 0.f;
    for (int i = 0; i < na.npreds; ++i) {
        float ww = 1.f / (1.f + expf(-edge_w[(size_t)na.node * max_deg + i]));
        const _Float16* ap = act_base + (size_t)na.preds[i] * act_stride + (size_t)p * CP;
        #pragma unroll
        for (int c = 0; c < CP; ++c) acc[c] += ww * (float)ap[c];
    }
    _Float16* o = h + (size_t)p * CP;
    #pragma unroll
    for (int c = 0; c < CP; ++c) o[c] = (_Float16)(acc[c] > 0.f ? acc[c] : 0.f);
}

// ---------------- inner conv: implicit GEMM, 3x3 stride-1, 32->32, WMMA f16 ----------------
// one wave = 16 positions x 32 out-channels; 9 taps -> 18 v_wmma per wave
__global__ void __launch_bounds__(256)
k_conv_inner(const _Float16* __restrict__ hbuf, const v16h* __restrict__ wpk,
             float* __restrict__ zbuf) {
    int wave = threadIdx.x >> 5, lane = threadIdx.x & 31;
    int tile = blockIdx.x * 8 + wave;                  // 4096 tiles of 16 positions
    int kg = lane >> 4, nn = lane & 15;
    int b = tile >> 8, rem = tile & 255;
    int y = rem >> 2, x0 = (rem & 3) << 4;
    int x = x0 + nn;                                   // A-row m = lane&15
    v8f acc0 = {}, acc1 = {};
    #pragma unroll
    for (int t = 0; t < 9; ++t) {
        int iy = y + t / 3 - 1, ix = x + t % 3 - 1;
        v16h a = {};
        if ((unsigned)iy < (unsigned)HWO && (unsigned)ix < (unsigned)HWO) {
            const char* base = (const char*)(hbuf + (((size_t)(b * HWO + iy)) * HWO + ix) * CP);
            v8h lo = *(const v8h*)(base + 16 * kg);        // ch 8kg..8kg+7
            v8h hi = *(const v8h*)(base + 32 + 16 * kg);   // ch 16+8kg..+7
            #pragma unroll
            for (int i = 0; i < 8; ++i) { a[i] = lo[i]; a[i + 8] = hi[i]; }
        }
        v16h b0 = wpk[(t * 2 + 0) * 32 + lane];
        v16h b1 = wpk[(t * 2 + 1) * 32 + lane];
        acc0 = __builtin_amdgcn_wmma_f32_16x16x32_f16(false, a, false, b0, (short)0, acc0, false, false);
        acc1 = __builtin_amdgcn_wmma_f32_16x16x32_f16(false, a, false, b1, (short)0, acc1, false, false);
    }
    float* zb = zbuf + (size_t)tile * 16 * CP;         // NHWC: pos-major, 32 ch
    #pragma unroll
    for (int r = 0; r < 8; ++r) {
        int m = kg * 8 + r;
        zb[m * CP + nn]      = acc0[r];
        zb[m * CP + 16 + nn] = acc1[r];
    }
}

// ---------------- input conv: 3x3 stride-2, 16->32; K padded 144->160 (5 chunks) ----------------
__global__ void __launch_bounds__(256)
k_conv_input(const _Float16* __restrict__ xr, const v16h* __restrict__ wpk,
             float* __restrict__ zbuf) {
    int wave = threadIdx.x >> 5, lane = threadIdx.x & 31;
    int tile = blockIdx.x * 8 + wave;
    int kg = lane >> 4, nn = lane & 15;
    int b = tile >> 8, rem = tile & 255;
    int y = rem >> 2, x0 = (rem & 3) << 4;
    int x = x0 + nn;
    v8f acc0 = {}, acc1 = {};
    #pragma unroll
    for (int t = 0; t < 5; ++t) {
        int t0 = 2 * t, t1 = 2 * t + 1;
        v8h lo = {}, hi = {};
        {
            int iy = 2 * y + t0 / 3 - 1, ix = 2 * x + t0 % 3 - 1;
            if ((unsigned)iy < (unsigned)HWI && (unsigned)ix < (unsigned)HWI)
                lo = *(const v8h*)((const char*)(xr + (((size_t)(b * HWI + iy)) * HWI + ix) * CIN) + 16 * kg);
        }
        if (t1 < 9) {
            int iy = 2 * y + t1 / 3 - 1, ix = 2 * x + t1 % 3 - 1;
            if ((unsigned)iy < (unsigned)HWI && (unsigned)ix < (unsigned)HWI)
                hi = *(const v8h*)((const char*)(xr + (((size_t)(b * HWI + iy)) * HWI + ix) * CIN) + 16 * kg);
        }
        v16h a;
        #pragma unroll
        for (int i = 0; i < 8; ++i) { a[i] = lo[i]; a[i + 8] = hi[i]; }
        v16h b0 = wpk[(t * 2 + 0) * 32 + lane];
        v16h b1 = wpk[(t * 2 + 1) * 32 + lane];
        acc0 = __builtin_amdgcn_wmma_f32_16x16x32_f16(false, a, false, b0, (short)0, acc0, false, false);
        acc1 = __builtin_amdgcn_wmma_f32_16x16x32_f16(false, a, false, b1, (short)0, acc1, false, false);
    }
    float* zb = zbuf + (size_t)tile * 16 * CP;
    #pragma unroll
    for (int r = 0; r < 8; ++r) {
        int m = kg * 8 + r;
        zb[m * CP + nn]      = acc0[r];
        zb[m * CP + 16 + nn] = acc1[r];
    }
}

// ---------------- BN stats: per-channel sum & sumsq over 65536 positions ----------------
__global__ void k_zero(float* p, int n) {
    int i = blockIdx.x * blockDim.x + threadIdx.x;
    if (i < n) p[i] = 0.f;
}

__global__ void k_stats(const float* __restrict__ z, float* __restrict__ stats) {
    __shared__ float ssum[256], ssq[256];
    int c = threadIdx.x & 31, g = threadIdx.x >> 5;     // 8 position-groups / block
    float s = 0.f, q = 0.f;
    for (int p = blockIdx.x * 8 + g; p < NPOS; p += gridDim.x * 8) {
        float v = z[(size_t)p * CP + c];
        s += v; q += v * v;
    }
    ssum[threadIdx.x] = s; ssq[threadIdx.x] = q;
    __syncthreads();
    if (g == 0) {
        #pragma unroll
        for (int k = 1; k < 8; ++k) { s += ssum[c + 32 * k]; q += ssq[c + 32 * k]; }
        atomicAdd(&stats[c], s);
        atomicAdd(&stats[32 + c], q);
    }
}

__global__ void k_finalize(const float* __restrict__ stats, const float* __restrict__ gamma,
                           const float* __restrict__ beta, int node, float* __restrict__ ss) {
    int c = threadIdx.x;
    if (c >= CP) return;
    const float Ninv = 1.f / (float)NPOS;
    float mu  = stats[c] * Ninv;
    float var = stats[32 + c] * Ninv - mu * mu;
    float rs  = rsqrtf(var + EPSBN);
    float sc  = gamma[(size_t)node * CP + c] * rs;
    ss[c]      = sc;
    ss[32 + c] = beta[(size_t)node * CP + c] - sc * mu;
}

// ---------------- BN apply + f16 act store + bottom-node output accumulate ----------------
__global__ void k_apply(const float* __restrict__ z, const float* __restrict__ ss,
                        _Float16* __restrict__ actout, float* __restrict__ out,
                        int isbottom, float invnb) {
    int p = blockIdx.x * blockDim.x + threadIdx.x;
    if (p >= NPOS) return;
    int b = p >> 12, y = (p >> 6) & 63, xx = p & 63;
    #pragma unroll
    for (int c = 0; c < CP; ++c) {
        float o = ss[c] * z[(size_t)p * CP + c] + ss[32 + c];
        actout[(size_t)p * CP + c] = (_Float16)o;
        if (isbottom)
            out[(((size_t)b * CP + c) * HWO + y) * HWO + xx] += o * invnb;
    }
}

// ================= host side =================
namespace {
struct MT19937 {
    uint32_t mt[624]; int mti;
    void seed(uint32_t s) {
        mt[0] = s;
        for (mti = 1; mti < 624; ++mti)
            mt[mti] = 1812433253u * (mt[mti - 1] ^ (mt[mti - 1] >> 30)) + (uint32_t)mti;
    }
    uint32_t next() {
        if (mti >= 624) {
            for (int k = 0; k < 624; ++k) {
                uint32_t y = (mt[k] & 0x80000000u) | (mt[(k + 1) % 624] & 0x7fffffffu);
                mt[k] = mt[(k + 397) % 624] ^ (y >> 1) ^ ((y & 1u) ? 0x9908b0dfu : 0u);
            }
            mti = 0;
        }
        uint32_t y = mt[mti++];
        y ^= y >> 11; y ^= (y << 7) & 0x9d2c5680u; y ^= (y << 15) & 0xefc60000u; y ^= y >> 18;
        return y;
    }
    double rnd() {  // numpy random_sample
        uint32_t a = next() >> 5, b = next() >> 6;
        return (a * 67108864.0 + b) / 9007199254740992.0;
    }
};
} // namespace

extern "C" void kernel_launch(void* const* d_in, const int* in_sizes, int n_in_args,
                              void* d_out, int out_size, void* d_ws, size_t ws_size,
                              hipStream_t stream) {
    const float* x_in       = (const float*)d_in[0];
    const float* conv_in_w  = (const float*)d_in[1];
    const float* conv_inn_w = (const float*)d_in[2];
    const float* gamma      = (const float*)d_in[3];
    const float* beta       = (const float*)d_in[4];
    const float* edge_w     = (const float*)d_in[5];
    float* out = (float*)d_out;

    // ---- rebuild the reference DAG (MT19937(42), p=0.15) ----
    int preds[NNODES][NNODES]; int npreds[NNODES];
    for (int j = 0; j < NNODES; ++j) npreds[j] = 0;
    MT19937 mt; mt.seed(42u);
    for (int j = 1; j < NNODES; ++j)
        for (int i = 0; i < j; ++i)
            if (mt.rnd() < 0.15) preds[j][npreds[j]++] = i;
    bool has_succ[NNODES] = {false};
    for (int j = 0; j < NNODES; ++j)
        for (int k = 0; k < npreds[j]; ++k) has_succ[preds[j][k]] = true;
    int in_idx[NNODES], inner_idx[NNODES];
    int n_in_nodes = 0, n_inner = 0, n_bottom = 0, max_deg = 1;
    bool is_bottom[NNODES];
    for (int n = 0; n < NNODES; ++n) {
        if (npreds[n] == 0) { in_idx[n] = n_in_nodes++; inner_idx[n] = -1; }
        else                { inner_idx[n] = n_inner++; in_idx[n] = -1; }
        if (npreds[n] > max_deg) max_deg = npreds[n];
        is_bottom[n] = !has_succ[n];
        if (is_bottom[n]) ++n_bottom;
    }
    float invnb = 1.f / (float)n_bottom;

    // ---- workspace layout (256B aligned slabs) ----
    char* ws = (char*)d_ws;
    size_t off = 0;
    auto alloc = [&](size_t bytes) { size_t o = off; off = (off + bytes + 255) & ~(size_t)255; return o; };
    const size_t ACT_ELEMS = (size_t)NPOS * CP;                      // per-node act, f16
    size_t o_xr    = alloc((size_t)NB * HWI * HWI * CIN * 2);
    size_t o_h     = alloc(ACT_ELEMS * 2);
    size_t o_z     = alloc(ACT_ELEMS * 4);
    size_t o_stats = alloc(64 * 4);
    size_t o_ss    = alloc(64 * 4);
    size_t o_wpin  = alloc((size_t)n_in_nodes * 5 * 2 * 32 * 16 * 2);
    size_t o_wpinn = alloc((size_t)n_inner   * 9 * 2 * 32 * 16 * 2);
    size_t o_act   = alloc((size_t)NNODES * ACT_ELEMS * 2);
    if (off > ws_size) return;   // workspace too small; nothing sane to do

    _Float16* xr      = (_Float16*)(ws + o_xr);
    _Float16* hbuf    = (_Float16*)(ws + o_h);
    float*    zbuf    = (float*)(ws + o_z);
    float*    statsb  = (float*)(ws + o_stats);
    float*    ssb     = (float*)(ws + o_ss);
    _Float16* wpin    = (_Float16*)(ws + o_wpin);
    _Float16* wpinn   = (_Float16*)(ws + o_wpinn);
    _Float16* actbase = (_Float16*)(ws + o_act);

    // ---- setup kernels ----
    k_zero<<<(out_size + 255) / 256, 256, 0, stream>>>(out, out_size);
    {
        int n = NB * HWI * HWI;
        k_prep_xr<<<(n + 255) / 256, 256, 0, stream>>>(x_in, xr);
    }
    {
        int tot = n_in_nodes * 5 * 2 * 32 * 16;
        k_pack_in<<<(tot + 255) / 256, 256, 0, stream>>>(conv_in_w, wpin, n_in_nodes);
    }
    {
        int tot = n_inner * 9 * 2 * 32 * 16;
        k_pack_inner<<<(tot + 255) / 256, 256, 0, stream>>>(conv_inn_w, wpinn, n_inner);
    }

    // ---- topological sweep (all edges i<j, so 0..31 is topo order) ----
    const int convBlocks = (NPOS / 16) / 8;   // 4096 tiles, 8 waves/block = 512
    for (int n = 0; n < NNODES; ++n) {
        if (npreds[n] == 0) {
            const v16h* wpk = (const v16h*)(wpin + (size_t)in_idx[n] * 5 * 2 * 32 * 16);
            k_conv_input<<<convBlocks, 256, 0, stream>>>(xr, wpk, zbuf);
        } else {
            NodeArgs na; na.npreds = npreds[n]; na.node = n;
            for (int k = 0; k < NNODES; ++k) na.preds[k] = (k < npreds[n]) ? preds[n][k] : 0;
            k_gather<<<(NPOS + 255) / 256, 256, 0, stream>>>(actbase, ACT_ELEMS, edge_w,
                                                             max_deg, na, hbuf);
            const v16h* wpk = (const v16h*)(wpinn + (size_t)inner_idx[n] * 9 * 2 * 32 * 16);
            k_conv_inner<<<convBlocks, 256, 0, stream>>>(hbuf, wpk, zbuf);
        }
        k_zero<<<1, 64, 0, stream>>>(statsb, 64);
        k_stats<<<64, 256, 0, stream>>>(zbuf, statsb);
        k_finalize<<<1, 32, 0, stream>>>(statsb, gamma, beta, n, ssb);
        k_apply<<<(NPOS + 255) / 256, 256, 0, stream>>>(zbuf, ssb,
                                                        actbase + (size_t)n * ACT_ELEMS,
                                                        out, is_bottom[n] ? 1 : 0, invnb);
    }
}
